// SeqDecoder_24120536335037
// MI455X (gfx1250) — matrix operand order repaired and checked by reading, not compiled, not run on hardware
//
#include <hip/hip_runtime.h>

#define D     512
#define NB    32
#define LSEQ  8
#define LM    512
#define EXT   32
#define NUMV  16
#define VOC   48
#define NEGF  (-3.4028234663852886e38f)

typedef float v2f __attribute__((ext_vector_type(2)));
typedef float v8f __attribute__((ext_vector_type(8)));
typedef __bf16 bf16;
typedef bf16 v8bf  __attribute__((ext_vector_type(8)));
typedef bf16 v16bf __attribute__((ext_vector_type(16)));

// ---------------- f32 WMMA (16x16x4): D = A(16x4)*B(4x16)+C ----------------
static __device__ __forceinline__ v8f wmma4(v2f a, v2f b, v8f c) {
  return __builtin_amdgcn_wmma_f32_16x16x4_f32(false, a, false, b, (short)0, c,
                                               false, false);
}

static __device__ __forceinline__ v2f load_a(const float* A, int lda, int row0,
                                             int k, int lane) {
  int m  = lane & 15;
  int ks = (lane >> 4) << 1;
  const float* p = A + (row0 + m) * lda + k + ks;
  v2f a; a.x = p[0]; a.y = p[1];
  return a;
}

static __device__ __forceinline__ v2f load_b(const float* W, int ldw, int col0,
                                             int k, int lane) {
  int nn = lane & 15;
  int ks = (lane >> 4) << 1;
  const float* p = W + (col0 + nn) * ldw + k + ks;
  v2f b; b.x = p[0]; b.y = p[1];
  return b;
}

// --------------- bf16 WMMA (16x16x32): D = A(16x32)*B(32x16)+C -------------
static __device__ __forceinline__ v8f wmma_bf(v16bf a, v16bf b, v8f c) {
  return __builtin_amdgcn_wmma_f32_16x16x32_bf16(false, a, false, b, (short)0,
                                                 c, false, false);
}

static __device__ __forceinline__ v16bf cat8(v8bf lo, v8bf hi) {
  return __builtin_shufflevector(lo, hi, 0, 1, 2, 3, 4, 5, 6, 7, 8, 9, 10, 11,
                                 12, 13, 14, 15);
}

// A 16x32 bf16 fragment: lane m=lane&15; VGPR 0-3: K=(lane>>4)*8..+7,
// VGPR 4-7: +16. Two 16B contiguous chunks.
static __device__ __forceinline__ v16bf load_a16(const bf16* A, int lda,
                                                 int row0, int k0, int lane) {
  int m  = lane & 15;
  int ks = (lane >> 4) * 8;
  const bf16* p = A + (size_t)(row0 + m) * lda + k0 + ks;
  v8bf lo = *(const v8bf*)p;
  v8bf hi = *(const v8bf*)(p + 16);
  return cat8(lo, hi);
}

// B 32x16 bf16 fragment from W[col][k] row-major: lane n=lane&15 holds
// K=(lane>>4)*8..+7 and +16. Two 16B contiguous chunks.
static __device__ __forceinline__ v16bf load_b16(const bf16* W, int ldw,
                                                 int col0, int k0, int lane) {
  int nn = lane & 15;
  int kb = (lane >> 4) * 8;
  const bf16* p = W + (size_t)(col0 + nn) * ldw + k0 + kb;
  v8bf lo = *(const v8bf*)p;
  v8bf hi = *(const v8bf*)(p + 16);
  return cat8(lo, hi);
}

// ---------------------------------------------------------------- embedding
__global__ void gather_emb(const int* __restrict__ ids,
                           const float* __restrict__ ext_emb,
                           const float* __restrict__ num_emb,
                           float* __restrict__ dec_in) {
  int row = blockIdx.x;  // n*LSEQ + l, 0..255
  int n   = row >> 3;
  int id  = ids[row];
  const float* src = (id < EXT) ? (ext_emb + id * D)
                                : (num_emb + (n * NUMV + (id - EXT)) * D);
  for (int d = threadIdx.x; d < D; d += blockDim.x)
    dec_in[row * D + d] = src[d];
}

// ------------------------------------------------------- f32 -> bf16 rows
// dst[row*dst_ld + k] = (bf16) src[row*src_ld + src_off + k]
__global__ void cvt_bf16_rows(const float* __restrict__ src, int src_ld,
                              int src_off, bf16* __restrict__ dst, int dst_ld,
                              int ncols) {
  int row = blockIdx.x;
  const float* s = src + (size_t)row * src_ld + src_off;
  bf16* d = dst + (size_t)row * dst_ld;
  for (int k = threadIdx.x; k < ncols; k += blockDim.x) d[k] = (bf16)s[k];
}

// ------------------------------------------------- generic f32 WMMA GEMM
// C[M x Ncols] = act( A1 @ W[woff1]^T (+ A2 @ W[woff2]^T) + bias )
// block = 256 threads (8 waves); block tile = 32 rows x 128 cols
__global__ void gemm_wmma(const float* __restrict__ A1,
                          const float* __restrict__ A2,
                          const float* __restrict__ W, int ldw, int woff1,
                          int woff2, const float* __restrict__ bias,
                          float* __restrict__ C, int Ncols, int K, int act) {
  int lane = threadIdx.x & 31;
  int wv   = threadIdx.x >> 5;
  int row0 = blockIdx.y * 32 + (wv >> 2) * 16;
  int colb = blockIdx.x * 128 + (wv & 3) * 32;
  v8f acc0 = {}; v8f acc1 = {};
  int npass = A2 ? 2 : 1;
  for (int pass = 0; pass < npass; ++pass) {
    const float* A  = pass ? A2 : A1;
    const float* Wp = W + (pass ? woff2 : woff1);
    for (int k = 0; k < K; k += 4) {
      v2f a  = load_a(A, K, row0, k, lane);
      v2f b0 = load_b(Wp, ldw, colb, k, lane);
      v2f b1 = load_b(Wp, ldw, colb + 16, k, lane);
      acc0 = wmma4(a, b0, acc0);
      acc1 = wmma4(a, b1, acc1);
    }
  }
  int mb = (lane & 16) ? 8 : 0;
  int nn = lane & 15;
#pragma unroll
  for (int v = 0; v < 8; ++v) {
    int r = row0 + mb + v;
    int c0 = colb + nn;
    float v0 = acc0[v] + bias[c0];
    int c1 = colb + 16 + nn;
    float v1 = acc1[v] + bias[c1];
    if (act) { v0 = tanhf(v0); v1 = tanhf(v1); }
    C[r * Ncols + c0] = v0;
    C[r * Ncols + c1] = v1;
  }
}

// -------------------------------------------------------------- GRU step
// block = 64 threads (2 waves, one per 16-row m-tile); grid.x = D/16 col tiles
__global__ void gru_step(const float* __restrict__ h_in,
                         const float* __restrict__ GX,
                         const float* __restrict__ Wh,
                         const float* __restrict__ bh, int t,
                         float* __restrict__ h_out, float* __restrict__ qs,
                         float* __restrict__ hn_out) {
  int lane = threadIdx.x & 31;
  int wv   = threadIdx.x >> 5;
  int row0 = wv * 16;
  int colb = blockIdx.x * 16;
  v8f acc0 = {}, acc1 = {}, acc2 = {};
  for (int k = 0; k < D; k += 4) {
    v2f a  = load_a(h_in, D, row0, k, lane);
    v2f b0 = load_b(Wh, D, colb, k, lane);
    v2f b1 = load_b(Wh, D, D + colb, k, lane);
    v2f b2 = load_b(Wh, D, 2 * D + colb, k, lane);
    acc0 = wmma4(a, b0, acc0);   // reset gate
    acc1 = wmma4(a, b1, acc1);   // update gate
    acc2 = wmma4(a, b2, acc2);   // new gate
  }
  int mb = (lane & 16) ? 8 : 0;
  int c  = colb + (lane & 15);
#pragma unroll
  for (int v = 0; v < 8; ++v) {
    int n = row0 + mb + v;  // batch row
    const float* gx = GX + (n * LSEQ + t) * (3 * D);
    float ghr = acc0[v] + bh[c];
    float ghz = acc1[v] + bh[D + c];
    float ghn = acc2[v] + bh[2 * D + c];
    float r   = 1.f / (1.f + expf(-(gx[c] + ghr)));
    float z   = 1.f / (1.f + expf(-(gx[D + c] + ghz)));
    float nn2 = tanhf(gx[2 * D + c] + r * ghn);
    float hold = h_in[n * D + c];
    float hnew = (1.f - z) * nn2 + z * hold;
    h_out[n * D + c] = hnew;
    qs[(n * LSEQ + t) * D + c] = hnew;
    if (hn_out) hn_out[n * D + c] = hnew;
  }
}

// ------------------------- fused m_proj (bf16 WMMA) + tanh scores ----------
// grid = (LM/16, NB); block = 256 threads (8 waves)
#define MP_STRIDE 516  // 512 + 4-float pad -> conflict-free mm-varying reads
__global__ void attn_score(const bf16* __restrict__ memB,
                           const bf16* __restrict__ WmB,
                           const float* __restrict__ q_proj,
                           const float* __restrict__ W1,
                           const float* __restrict__ b1,
                           const unsigned char* __restrict__ cmask,
                           float* __restrict__ wbuf) {
  __shared__ float mp[16 * MP_STRIDE];
  __shared__ float red[256];
  int n  = blockIdx.y;
  int mt = blockIdx.x;
  int lane = threadIdx.x & 31;
  int wv   = threadIdx.x >> 5;  // 0..7 -> 64-col slice of m_proj tile
  const bf16* A = memB + (size_t)(n * LM + mt * 16) * D;
  int colb = wv * 64;
  v8f acc[4] = {{}, {}, {}, {}};
  for (int k = 0; k < D; k += 32) {
    v16bf a = load_a16(A, D, 0, k, lane);
#pragma unroll
    for (int tt = 0; tt < 4; ++tt) {
      v16bf b = load_b16(WmB, D, colb + tt * 16, k, lane);
      acc[tt] = wmma_bf(a, b, acc[tt]);
    }
  }
  int mb = (lane & 16) ? 8 : 0;
  int nn = lane & 15;
#pragma unroll
  for (int tt = 0; tt < 4; ++tt)
#pragma unroll
    for (int v = 0; v < 8; ++v)
      mp[(mb + v) * MP_STRIDE + colb + tt * 16 + nn] = acc[tt][v];
  __syncthreads();

  // scores: 128 (l,mm) pairs, 2 threads each (d halves of 256)
  int tid  = threadIdx.x;
  int p    = tid >> 1;
  int half = tid & 1;
  int l    = p >> 4;
  int mm   = p & 15;
  const float* qph  = q_proj + (size_t)(n * LSEQ + l) * D + half * 256;
  const float* mrow = mp + mm * MP_STRIDE + half * 256;
  const float* w1h  = W1 + half * 256;
  float s = 0.f;
  for (int d = 0; d < 256; ++d)
    s += tanhf(qph[d] + mrow[d]) * w1h[d];
  red[tid] = s;
  __syncthreads();
  if (half == 0) {
    int mg = mt * 16 + mm;
    float val = red[tid] + red[tid + 1] + b1[0];
    if (!cmask[(size_t)(n * LSEQ + l) * LM + mg]) val = NEGF;
    wbuf[(size_t)(n * LSEQ + l) * LM + mg] = val;
  }
}

// ------------------------------------------------------- softmax over Lm
__global__ void softmax_row(float* __restrict__ wbuf) {
  __shared__ float red[256];
  int row = blockIdx.x;
  float* p = wbuf + (size_t)row * LM;
  int tid = threadIdx.x;
  float a = p[tid], b = p[tid + 256];
  red[tid] = fmaxf(a, b);
  __syncthreads();
  for (int s = 128; s > 0; s >>= 1) {
    if (tid < s) red[tid] = fmaxf(red[tid], red[tid + s]);
    __syncthreads();
  }
  float mx = red[0];
  __syncthreads();
  float ea = expf(a - mx), eb = expf(b - mx);
  red[tid] = ea + eb;
  __syncthreads();
  for (int s = 128; s > 0; s >>= 1) {
    if (tid < s) red[tid] += red[tid + s];
    __syncthreads();
  }
  float inv = 1.f / red[0];
  p[tid]       = ea * inv;
  p[tid + 256] = eb * inv;
}

// --------------------------------------------------- context cs = w @ memory
// grid = (D/64, NB); block = 64
__global__ void context_k(const float* __restrict__ wbuf,
                          const float* __restrict__ memory,
                          float* __restrict__ cs) {
  int n = blockIdx.y;
  int d = blockIdx.x * 64 + threadIdx.x;
  float acc[LSEQ] = {};
  const float* mem = memory + (size_t)n * LM * D + d;
  const float* wr  = wbuf + (size_t)n * LSEQ * LM;
  for (int m = 0; m < LM; ++m) {
    float mv = mem[(size_t)m * D];
#pragma unroll
    for (int l = 0; l < LSEQ; ++l) acc[l] += wr[l * LM + m] * mv;
  }
  for (int l = 0; l < LSEQ; ++l) cs[(size_t)(n * LSEQ + l) * D + d] = acc[l];
}

// -------------------------------------------------------------- logits
__global__ void logits_k(const float* __restrict__ feat,
                         const float* __restrict__ ext_out,
                         const float* __restrict__ num_emb,
                         const unsigned char* __restrict__ vmask,
                         float* __restrict__ out) {
  int n = blockIdx.x;
  for (int p = threadIdx.x; p < LSEQ * VOC; p += blockDim.x) {
    int l = p / VOC, v = p % VOC;
    const float* f = feat + (size_t)(n * LSEQ + l) * D;
    const float* e = (v < EXT) ? (ext_out + v * D)
                               : (num_emb + (size_t)(n * NUMV + (v - EXT)) * D);
    float s = 0.f;
    for (int d = 0; d < D; ++d) s += f[d] * e[d];
    if (!vmask[n * VOC + v]) s = NEGF;
    out[(size_t)(n * LSEQ + l) * VOC + v] = s;
  }
}

extern "C" void kernel_launch(void* const* d_in, const int* in_sizes, int n_in,
                              void* d_out, int out_size, void* d_ws,
                              size_t ws_size, hipStream_t stream) {
  (void)in_sizes; (void)n_in; (void)out_size; (void)ws_size;
  const int*           ids     = (const int*)d_in[0];
  const float*         past    = (const float*)d_in[1];
  const float*         num_emb = (const float*)d_in[2];
  const unsigned char* vmask   = (const unsigned char*)d_in[3];
  const float*         memory  = (const float*)d_in[4];
  const unsigned char* cmask   = (const unsigned char*)d_in[5];
  const float*         ext_emb = (const float*)d_in[6];
  const float*         ext_out = (const float*)d_in[7];
  const float*         Wx      = (const float*)d_in[8];
  const float*         Wh      = (const float*)d_in[9];
  const float*         bx      = (const float*)d_in[10];
  const float*         bh      = (const float*)d_in[11];
  const float*         W0      = (const float*)d_in[12];
  const float*         b0      = (const float*)d_in[13];
  const float*         W1      = (const float*)d_in[14];
  const float*         b1      = (const float*)d_in[15];
  const float*         fcW     = (const float*)d_in[16];
  const float*         fcb     = (const float*)d_in[17];

  float* out = (float*)d_out;
  float* logits_out = out;
  float* hn_out     = out + NB * LSEQ * VOC;  // 12288

  float* ws     = (float*)d_ws;
  float* dec_in = ws;                 // 256*512
  float* GX     = dec_in + 131072;    // 256*1536
  float* h0     = GX + 393216;        // 32*512
  float* h1     = h0 + 16384;
  float* qs     = h1 + 16384;         // 256*512
  float* qproj  = qs + 131072;
  float* wbuf   = qproj + 131072;
  float* cs     = wbuf + 131072;
  float* feat   = cs + 131072;
  bf16*  memB   = (bf16*)(feat + 131072);          // 32*512*512 bf16 (16.8MB)
  bf16*  WmB    = (bf16*)((float*)memB + 2097152); // 512*512 bf16 (0.5MB)

  // 0) one-time f32->bf16 copies for the big GEMM operands
  cvt_bf16_rows<<<NB * LM, 256, 0, stream>>>(memory, D, 0, memB, D, D);
  cvt_bf16_rows<<<D, 256, 0, stream>>>(W0, 2 * D, D, WmB, D, D);
  // 1) embedding gather
  gather_emb<<<NB * LSEQ, 256, 0, stream>>>(ids, ext_emb, num_emb, dec_in);
  // 2) GX = dec_in @ Wx^T + bx   [256 x 1536]
  gemm_wmma<<<dim3(12, 8), 256, 0, stream>>>(dec_in, nullptr, Wx, D, 0, 0, bx,
                                             GX, 3 * D, D, 0);
  // 3) GRU recurrence (8 steps, ping-pong h)
  for (int t = 0; t < LSEQ; ++t) {
    const float* hin = (t == 0) ? past : ((t & 1) ? h0 : h1);
    float* hout      = (t & 1) ? h1 : h0;
    gru_step<<<D / 16, 64, 0, stream>>>(hin, GX, Wh, bh, t, hout, qs,
                                        (t == LSEQ - 1) ? hn_out : nullptr);
  }
  // 4) q_proj = qs @ attn_W0[:, :D]^T + b0   [256 x 512]
  gemm_wmma<<<dim3(4, 8), 256, 0, stream>>>(qs, nullptr, W0, 2 * D, 0, 0, b0,
                                            qproj, D, D, 0);
  // 5) fused m_proj (bf16 WMMA) + tanh scores (+mask)
  attn_score<<<dim3(LM / 16, NB), 256, 0, stream>>>(memB, WmB, qproj, W1, b1,
                                                    cmask, wbuf);
  // 6) softmax over Lm
  softmax_row<<<NB * LSEQ, 256, 0, stream>>>(wbuf);
  // 7) context
  context_k<<<dim3(D / 64, NB), 64, 0, stream>>>(wbuf, memory, cs);
  // 8) feat = tanh(qs@fcW[:,:D]^T + cs@fcW[:,D:]^T + fcb)
  gemm_wmma<<<dim3(4, 8), 256, 0, stream>>>(qs, cs, fcW, 2 * D, 0, D, fcb,
                                            feat, D, D, 1);
  // 9) logits + vocab mask
  logits_k<<<NB, 256, 0, stream>>>(feat, ext_out, num_emb, vmask, logits_out);
}
